// Bottleneck_23218593202473
// MI455X (gfx1250) — 8-bit matrix operand order repaired and checked by reading, compiled, not run on hardware
//
#include <hip/hip_runtime.h>
#include <hip/hip_bf16.h>
#include <stdint.h>

typedef int   v16i __attribute__((ext_vector_type(16)));
typedef float v8f  __attribute__((ext_vector_type(8)));

#define EPSC  1e-5f
#define Bn    64
#define CIN   1024
#define PCH   256
#define COUT  1024
#define HWsz  196
#define MTOT  (Bn*HWsz)
#define MT    128
#define NT    128
#define KT    128
#define ASTR  144
#define BSTR  136

__device__ __forceinline__ unsigned char sgnf8(float y) {
    return y > 0.f ? 0x38u : (y < 0.f ? 0xB8u : 0x00u);
}

__device__ __forceinline__ v16i ld_fragA(const unsigned char* As, int row0) {
    int lane = threadIdx.x & 31;
    const unsigned char* p = As + (size_t)(row0 + (lane & 15)) * ASTR + ((lane >> 4) << 3);
    v16i f;
#pragma unroll
    for (int half = 0; half < 2; ++half)
#pragma unroll
        for (int g = 0; g < 4; ++g) {
            uint64_t d = *(const uint64_t*)(const void*)(p + half * 64 + g * 16);
            f[half*8 + 2*g]     = (int)(uint32_t)d;
            f[half*8 + 2*g + 1] = (int)(uint32_t)(d >> 32);
        }
    return f;
}
__device__ __forceinline__ v16i ld_fragB(const unsigned char* Bs, int col0) {
    int lane = threadIdx.x & 31;
    v16i f;
    const unsigned char* p = Bs + (size_t)((lane >> 4) << 3) * BSTR + col0 + (lane & 15);
#pragma unroll
    for (int r = 0; r < 16; ++r) {
        uint32_t v = 0;
#pragma unroll
        for (int bb = 0; bb < 4; ++bb) {
            int k = (r >> 3) * 64 + ((r >> 1) & 3) * 16 + (r & 1) * 4 + bb;
            v |= (uint32_t)p[(size_t)k * BSTR] << (8 * bb);
        }
        f[r] = (int)v;
    }
    return f;
}
__device__ __forceinline__ void st_lds16(unsigned char* dst, int4 v) {
    uint64_t lo = ((uint64_t)(uint32_t)v.y << 32) | (uint32_t)v.x;
    uint64_t hi = ((uint64_t)(uint32_t)v.w << 32) | (uint32_t)v.z;
    *(uint64_t*)(void*)dst       = lo;
    *(uint64_t*)(void*)(dst + 8) = hi;
}

__global__ void k_sign_x(const float* __restrict__ x, unsigned char* __restrict__ a0) {
    size_t i = (size_t)blockIdx.x * blockDim.x + threadIdx.x;
    if (i >= (size_t)Bn*CIN*HWsz) return;
    int hw = (int)(i % HWsz);
    size_t bc = i / HWsz;
    int c = (int)(bc % CIN);
    int b = (int)(bc / CIN);
    a0[((size_t)b*HWsz + hw)*CIN + c] = sgnf8(x[i]);
}
__global__ void k_sign_w1t(const float* __restrict__ w1, unsigned char* __restrict__ o) {
    int i = blockIdx.x*blockDim.x + threadIdx.x;
    if (i >= CIN*PCH) return;
    int n = i & (PCH-1), k = i >> 8;
    o[i] = sgnf8(w1[(size_t)n*CIN + k]);
}
__global__ void k_sign_w2t(const float* __restrict__ w2, unsigned char* __restrict__ o) {
    int i = blockIdx.x*blockDim.x + threadIdx.x;
    if (i >= 9*PCH*PCH) return;
    int n = i & 255, k = (i >> 8) & 255, tp = i >> 16;
    o[i] = sgnf8(w2[((size_t)n*PCH + k)*9 + tp]);
}
__global__ void k_sign_w3t(const float* __restrict__ w3, unsigned char* __restrict__ o) {
    int i = blockIdx.x*blockDim.x + threadIdx.x;
    if (i >= PCH*COUT) return;
    int n = i & (COUT-1), k = i >> 10;
    o[i] = sgnf8(w3[(size_t)n*PCH + k]);
}
__global__ void k_bnp(const float* __restrict__ g, const float* __restrict__ b,
                      const float* __restrict__ m, const float* __restrict__ v,
                      float* __restrict__ scale, float* __restrict__ shift, int n) {
    int i = blockIdx.x*blockDim.x + threadIdx.x;
    if (i >= n) return;
    float r = rsqrtf(v[i] + EPSC);
    float s = g[i] * r;
    scale[i] = s;
    shift[i] = b[i] - m[i]*s;
}


__global__ __launch_bounds__(256) void k_gemm1(
    const unsigned char* __restrict__ A0, const unsigned char* __restrict__ W1s,
    const float* __restrict__ scale, const float* __restrict__ shift,
    unsigned char* __restrict__ act2)
{
    __shared__ __align__(16) unsigned char As[MT*ASTR];
    __shared__ __align__(16) unsigned char Bs[KT*BSTR];
    const int mt = blockIdx.x % (MTOT/MT);
    const int nt = blockIdx.x / (MTOT/MT);
    const int mbase = mt*MT, nbase = nt*NT;
    const int t = threadIdx.x;
    const int wid = t >> 5, wm = wid & 3, wn = wid >> 2;
    const int ar = t >> 1, aseg = (t & 1)*64;
    const int br = t >> 1, bseg = (t & 1)*64;

    v8f zero = {0.f,0.f,0.f,0.f,0.f,0.f,0.f,0.f};
    v8f acc[2][4];
#pragma unroll
    for (int i = 0; i < 2; ++i)
#pragma unroll
        for (int j = 0; j < 4; ++j) acc[i][j] = zero;

    for (int k0 = 0; k0 < CIN; k0 += KT) {
        const int4* ga = (const int4*)(const void*)(A0 + (size_t)(mbase+ar)*CIN + k0 + aseg);
        const int4* gb = (const int4*)(const void*)(W1s + (size_t)(k0+br)*PCH + nbase + bseg);
        int4 av[4], bv[4];
#pragma unroll
        for (int s = 0; s < 4; ++s) { av[s] = ga[s]; bv[s] = gb[s]; }
#pragma unroll
        for (int s = 0; s < 4; ++s) {
            *(int4*)(void*)(As + (size_t)ar*ASTR + aseg + s*16) = av[s];
            st_lds16(Bs + (size_t)br*BSTR + bseg + s*16, bv[s]);
        }
        __syncthreads();
        v16i af[2], bf[4];
#pragma unroll
        for (int i = 0; i < 2; ++i) af[i] = ld_fragA(As, wm*32 + i*16);
#pragma unroll
        for (int j = 0; j < 4; ++j) bf[j] = ld_fragB(Bs, wn*64 + j*16);
#pragma unroll
        for (int i = 0; i < 2; ++i)
#pragma unroll
            for (int j = 0; j < 4; ++j)
                acc[i][j] = __builtin_amdgcn_wmma_f32_16x16x128_fp8_fp8(
                    af[i], bf[j], (short)0, acc[i][j], false, false);
        __syncthreads();
    }
    const int lane = t & 31;
    const int cn = lane & 15, rsel = (lane >> 4) * 8;
#pragma unroll
    for (int j = 0; j < 4; ++j) {
        int n = nbase + wn*64 + j*16 + cn;
        float sc = scale[n], sh = shift[n];
#pragma unroll
        for (int i = 0; i < 2; ++i)
#pragma unroll
            for (int r = 0; r < 8; ++r) {
                int m = mbase + wm*32 + i*16 + rsel + r;
                act2[(size_t)m*PCH + n] = sgnf8(acc[i][j][r]*sc + sh);
            }
    }
}

__global__ __launch_bounds__(256) void k_gemm2(
    const unsigned char* __restrict__ act2, const unsigned char* __restrict__ W2s,
    const float* __restrict__ scale, const float* __restrict__ shift,
    unsigned char* __restrict__ act3)
{
    __shared__ __align__(16) unsigned char As[MT*ASTR];
    __shared__ __align__(16) unsigned char Bs[KT*BSTR];
    const int mt = blockIdx.x % (MTOT/MT);
    const int nt = blockIdx.x / (MTOT/MT);
    const int mbase = mt*MT, nbase = nt*NT;
    const int t = threadIdx.x;
    const int wid = t >> 5, wm = wid & 3, wn = wid >> 2;
    const int ar = t >> 1, aseg = (t & 1)*64;
    const int br = t >> 1, bseg = (t & 1)*64;

    const int mrow = mbase + ar;
    const int bidx = mrow / HWsz;
    const int hw   = mrow - bidx*HWsz;
    const int h    = hw / 14, w = hw - h*14;

    v8f zero = {0.f,0.f,0.f,0.f,0.f,0.f,0.f,0.f};
    v8f acc[2][4];
#pragma unroll
    for (int i = 0; i < 2; ++i)
#pragma unroll
        for (int j = 0; j < 4; ++j) acc[i][j] = zero;

    for (int tap = 0; tap < 9; ++tap) {
        const int dh = tap/3 - 1, dw = tap%3 - 1;
        const int h2 = h + dh, w2 = w + dw;
        const bool ok = ((unsigned)h2 < 14u) && ((unsigned)w2 < 14u);
        const size_t srow = ok ? (size_t)(bidx*HWsz + h2*14 + w2)*PCH : 0;
        for (int kc = 0; kc < PCH; kc += KT) {
            int4 av[4], bv[4];
#pragma unroll
            for (int s = 0; s < 4; ++s) av[s] = make_int4(0,0,0,0);
            if (ok) {
                const int4* ga = (const int4*)(const void*)(act2 + srow + kc + aseg);
#pragma unroll
                for (int s = 0; s < 4; ++s) av[s] = ga[s];
            }
            const int4* gb = (const int4*)(const void*)
                (W2s + (size_t)tap*PCH*PCH + (size_t)(kc+br)*PCH + nbase + bseg);
#pragma unroll
            for (int s = 0; s < 4; ++s) bv[s] = gb[s];
#pragma unroll
            for (int s = 0; s < 4; ++s) {
                *(int4*)(void*)(As + (size_t)ar*ASTR + aseg + s*16) = av[s];
                st_lds16(Bs + (size_t)br*BSTR + bseg + s*16, bv[s]);
            }
            __syncthreads();
            v16i af[2], bf[4];
#pragma unroll
            for (int i = 0; i < 2; ++i) af[i] = ld_fragA(As, wm*32 + i*16);
#pragma unroll
            for (int j = 0; j < 4; ++j) bf[j] = ld_fragB(Bs, wn*64 + j*16);
#pragma unroll
            for (int i = 0; i < 2; ++i)
#pragma unroll
                for (int j = 0; j < 4; ++j)
                    acc[i][j] = __builtin_amdgcn_wmma_f32_16x16x128_fp8_fp8(
                        af[i], bf[j], (short)0, acc[i][j], false, false);
            __syncthreads();
        }
    }
    const int lane = t & 31;
    const int cn = lane & 15, rsel = (lane >> 4) * 8;
#pragma unroll
    for (int j = 0; j < 4; ++j) {
        int n = nbase + wn*64 + j*16 + cn;
        float sc = scale[n], sh = shift[n];
#pragma unroll
        for (int i = 0; i < 2; ++i)
#pragma unroll
            for (int r = 0; r < 8; ++r) {
                int m = mbase + wm*32 + i*16 + rsel + r;
                act3[(size_t)m*PCH + n] = sgnf8(acc[i][j][r]*sc + sh);
            }
    }
}

__global__ __launch_bounds__(256) void k_gemm3(
    const unsigned char* __restrict__ act3, const unsigned char* __restrict__ W3s,
    const float* __restrict__ scale, const float* __restrict__ shift,
    const float* __restrict__ x, float* __restrict__ out)
{
    __shared__ __align__(16) unsigned char As[MT*ASTR];
    __shared__ __align__(16) unsigned char Bs[KT*BSTR];
    const int mt = blockIdx.x % (MTOT/MT);
    const int nt = blockIdx.x / (MTOT/MT);
    const int mbase = mt*MT, nbase = nt*NT;
    const int t = threadIdx.x;
    const int wid = t >> 5, wm = wid & 3, wn = wid >> 2;
    const int ar = t >> 1, aseg = (t & 1)*64;
    const int br = t >> 1, bseg = (t & 1)*64;

    v8f zero = {0.f,0.f,0.f,0.f,0.f,0.f,0.f,0.f};
    v8f acc[2][4];
#pragma unroll
    for (int i = 0; i < 2; ++i)
#pragma unroll
        for (int j = 0; j < 4; ++j) acc[i][j] = zero;

    for (int k0 = 0; k0 < PCH; k0 += KT) {
        const int4* ga = (const int4*)(const void*)(act3 + (size_t)(mbase+ar)*PCH + k0 + aseg);
        const int4* gb = (const int4*)(const void*)(W3s + (size_t)(k0+br)*COUT + nbase + bseg);
        int4 av[4], bv[4];
#pragma unroll
        for (int s = 0; s < 4; ++s) { av[s] = ga[s]; bv[s] = gb[s]; }
#pragma unroll
        for (int s = 0; s < 4; ++s) {
            *(int4*)(void*)(As + (size_t)ar*ASTR + aseg + s*16) = av[s];
            st_lds16(Bs + (size_t)br*BSTR + bseg + s*16, bv[s]);
        }
        __syncthreads();
        v16i af[2], bf[4];
#pragma unroll
        for (int i = 0; i < 2; ++i) af[i] = ld_fragA(As, wm*32 + i*16);
#pragma unroll
        for (int j = 0; j < 4; ++j) bf[j] = ld_fragB(Bs, wn*64 + j*16);
#pragma unroll
        for (int i = 0; i < 2; ++i)
#pragma unroll
            for (int j = 0; j < 4; ++j)
                acc[i][j] = __builtin_amdgcn_wmma_f32_16x16x128_fp8_fp8(
                    af[i], bf[j], (short)0, acc[i][j], false, false);
        __syncthreads();
    }
    const int lane = t & 31;
    const int cn = lane & 15, rsel = (lane >> 4) * 8;
#pragma unroll
    for (int j = 0; j < 4; ++j) {
        int n = nbase + wn*64 + j*16 + cn;
        float sc = scale[n], sh = shift[n];
#pragma unroll
        for (int i = 0; i < 2; ++i)
#pragma unroll
            for (int r = 0; r < 8; ++r) {
                int m = mbase + wm*32 + i*16 + rsel + r;
                int b = m / HWsz;
                int hw = m - b*HWsz;
                size_t oi = ((size_t)b*COUT + n)*HWsz + hw;
                out[oi] = acc[i][j][r]*sc + sh + x[oi];
            }
    }
}

extern "C" void kernel_launch(void* const* d_in, const int* in_sizes, int n_in,
                              void* d_out, int out_size, void* d_ws, size_t ws_size,
                              hipStream_t stream) {
    const float* x  = (const float*)d_in[0];
    const float* w1 = (const float*)d_in[1];
    const float* g1 = (const float*)d_in[2];
    const float* b1 = (const float*)d_in[3];
    const float* m1 = (const float*)d_in[4];
    const float* v1 = (const float*)d_in[5];
    const float* w2 = (const float*)d_in[6];
    const float* g2 = (const float*)d_in[7];
    const float* b2 = (const float*)d_in[8];
    const float* m2 = (const float*)d_in[9];
    const float* v2 = (const float*)d_in[10];
    const float* w3 = (const float*)d_in[11];
    const float* g3 = (const float*)d_in[12];
    const float* b3 = (const float*)d_in[13];
    const float* m3 = (const float*)d_in[14];
    const float* v3 = (const float*)d_in[15];

    char* ws = (char*)d_ws;
    unsigned char* A0   = (unsigned char*)(ws);
    unsigned char* act2 = (unsigned char*)(ws + 12845056);
    unsigned char* act3 = (unsigned char*)(ws + 16056320);
    unsigned char* W1s  = (unsigned char*)(ws + 19267584);
    unsigned char* W2s  = (unsigned char*)(ws + 19529728);
    unsigned char* W3s  = (unsigned char*)(ws + 20119552);
    float* sc1 = (float*)(ws + 20381696);
    float* sh1 = sc1 + 256;
    float* sc2 = sh1 + 256;
    float* sh2 = sc2 + 256;
    float* sc3 = sh2 + 256;
    float* sh3 = sc3 + 1024;

    k_sign_x  <<<50176, 256, 0, stream>>>(x, A0);
    k_sign_w1t<<<CIN*PCH/256, 256, 0, stream>>>(w1, W1s);
    k_sign_w2t<<<9*PCH*PCH/256, 256, 0, stream>>>(w2, W2s);
    k_sign_w3t<<<PCH*COUT/256, 256, 0, stream>>>(w3, W3s);
    k_bnp<<<1, 256, 0, stream>>>(g1, b1, m1, v1, sc1, sh1, PCH);
    k_bnp<<<1, 256, 0, stream>>>(g2, b2, m2, v2, sc2, sh2, PCH);
    k_bnp<<<4, 256, 0, stream>>>(g3, b3, m3, v3, sc3, sh3, COUT);

    k_gemm1<<<(MTOT/MT) * (PCH/NT),  256, 0, stream>>>(A0, W1s, sc1, sh1, act2);
    k_gemm2<<<(MTOT/MT) * (PCH/NT),  256, 0, stream>>>(act2, W2s, sc2, sh2, act3);
    k_gemm3<<<(MTOT/MT) * (COUT/NT), 256, 0, stream>>>(act3, W3s, sc3, sh3, x, (float*)d_out);
}
